// MultiHeadGATLayer_25598005084424
// MI455X (gfx1250) — matrix operand order repaired and checked by reading, not compiled, not run on hardware
//
#include <hip/hip_runtime.h>

// ---------------- problem constants (match reference) ----------------
constexpr int kNSent = 100000;
constexpr int kNType = 10000;
constexpr int kEdges = 400000;
constexpr int kOut   = 256;
constexpr int kH     = 8;
constexpr int kHD    = 32;          // kOut / kH
constexpr float kNegSlope = 0.01f;

typedef __attribute__((ext_vector_type(16))) _Float16 v16h;
typedef __attribute__((ext_vector_type(8)))  float    v8f;

// float atomic max via signed/unsigned int ordering trick (init must be -inf)
__device__ inline void atomicMaxFloat(float* addr, float val) {
    if (val >= 0.0f) atomicMax((int*)addr, __float_as_int(val));
    else             atomicMin((unsigned int*)addr, __float_as_uint(val));
}

// ---------------- pass 0: init scratch ----------------
__global__ void k_init(float* m_buf, float* den_buf, float* h_accum) {
    int t = blockIdx.x * blockDim.x + threadIdx.x;
    if (t < kNType * kH) {
        ((unsigned int*)m_buf)[t] = 0xFF800000u;  // -inf
        den_buf[t] = 0.0f;
    }
    if (t < kNType * kOut) h_accum[t] = 0.0f;
}

// ---------------- pass 1: per-(edge,head) attention logits + segment max ----
__global__ void k_logits(const float* __restrict__ h_sent,
                         const float* __restrict__ h_type,
                         const float* __restrict__ attn_w,
                         const int*   __restrict__ src,
                         const int*   __restrict__ dst,
                         float* __restrict__ logit,
                         float* __restrict__ m_buf) {
    __shared__ float s_attn[kH * 2 * kHD];   // 8 x 64
    for (int i = threadIdx.x; i < kH * 2 * kHD; i += blockDim.x)
        s_attn[i] = attn_w[i];
    __syncthreads();

    long long t = (long long)blockIdx.x * blockDim.x + threadIdx.x;
    if (t >= (long long)kEdges * kH) return;
    int e = (int)(t >> 3);
    int h = (int)(t & 7);
    int s = src[e];
    int d = dst[e];
    // flat-concat head slicing: heads 0-3 read src features, heads 4-7 dst features
    const float* base = (h < 4) ? (h_sent + (size_t)s * kOut + h * 64)
                                : (h_type + (size_t)d * kOut + (h - 4) * 64);
    const float* w = s_attn + h * 64;
    float acc = 0.0f;
#pragma unroll 8
    for (int k = 0; k < 64; ++k) acc += base[k] * w[k];
    acc = (acc > 0.0f) ? acc : kNegSlope * acc;   // leaky relu
    logit[t] = acc;
    atomicMaxFloat(m_buf + (size_t)d * kH + h, acc);
}

// ---------------- pass 2: exp(e - m) and segment sum -------------------
__global__ void k_expsum(const int* __restrict__ dst,
                         const float* __restrict__ m_buf,
                         float* __restrict__ exbuf,    // in: logits, out: exp
                         float* __restrict__ den_buf) {
    long long t = (long long)blockIdx.x * blockDim.x + threadIdx.x;
    if (t >= (long long)kEdges * kH) return;
    int e = (int)(t >> 3);
    int h = (int)(t & 7);
    int d = dst[e];
    float ex = expf(exbuf[t] - m_buf[(size_t)d * kH + h]);
    exbuf[t] = ex;
    unsafeAtomicAdd(den_buf + (size_t)d * kH + h, ex);
}

// ---------------- pass 3: alpha-weighted scatter of src features -------
__global__ void k_scatter(const float* __restrict__ h_sent,
                          const int*   __restrict__ src,
                          const int*   __restrict__ dst,
                          const float* __restrict__ exbuf,
                          const float* __restrict__ den_buf,
                          float* __restrict__ h_accum) {
    long long t = (long long)blockIdx.x * blockDim.x + threadIdx.x;
    if (t >= (long long)kEdges * kOut) return;
    int e  = (int)(t >> 8);
    int f  = (int)(t & 255);          // feature 0..255
    int h  = f >> 5;                  // head = f / 32
    int dn = dst[e];
    int sn = src[e];
    float ex  = exbuf[(size_t)e * kH + h];
    float den = den_buf[(size_t)dn * kH + h];
    float alpha = ex / fmaxf(den, 1e-9f);
    float val = alpha * h_sent[(size_t)sn * kOut + f];
    unsafeAtomicAdd(h_accum + (size_t)dn * kOut + f, val);
}

// ---------------- pass 4: f32 -> f16 conversion ------------------------
__global__ void k_cvt(const float* __restrict__ h_accum,
                      const float* __restrict__ w_comb,
                      _Float16* __restrict__ h16,
                      _Float16* __restrict__ w16) {
    int t = blockIdx.x * blockDim.x + threadIdx.x;
    if (t < kOut * kOut)    w16[t] = (_Float16)w_comb[t];
    if (t < kNType * kOut)  h16[t] = (_Float16)h_accum[t];
}

// ---------------- pass 5: WMMA GEMM  out = h16 @ w16^T + b ------------
// One wave per 16x16 output tile. 625 row-tiles x 16 col-tiles = 10000 tiles.
// K = 256 consumed in 8 steps of v_wmma_f32_16x16x32_f16.
__global__ void k_gemm(const _Float16* __restrict__ hA,   // (10000,256) row-major
                       const _Float16* __restrict__ wB,   // (256,256): row j = out col j, fast dim k
                       const float* __restrict__ b_comb,
                       float* __restrict__ out) {
    const int wave = threadIdx.x >> 5;
    const int lane = threadIdx.x & 31;
    const int tile = blockIdx.x * 8 + wave;       // wave-uniform
    if (tile >= (kNType / 16) * 16) return;       // uniform branch, EXEC stays all-1s
    const int tm = tile >> 4;     // row tile 0..624
    const int tn = tile & 15;     // col tile 0..15
    const int half = lane >> 4;   // 0: lanes 0-15, 1: lanes 16-31
    const int l16  = lane & 15;

    // A: lane holds row M=l16; K halves split by lane half.
    const _Float16* arow = hA + (size_t)(tm * 16 + l16) * kOut;
    // B: lane holds column N=l16; K split by lane half (16 each).
    const _Float16* brow = wB + (size_t)(tn * 16 + l16) * kOut;

    v8f acc = {};
#pragma unroll
    for (int k0 = 0; k0 < kOut; k0 += 32) {
        v16h a, b;
        // A 16-bit 16x32 layout: a[0..7] -> K = half*8 + 0..7 ; a[8..15] -> K = 16 + half*8 + 0..7
#pragma unroll
        for (int j = 0; j < 8; ++j) {
            a[j]     = arow[k0 + half * 8 + j];
            a[8 + j] = arow[k0 + 16 + half * 8 + j];
        }
        // B 16-bit 32x16 layout: b[0..7] -> K = half*8 + 0..7 ; b[8..15] -> K = 16 + half*8 + 0..7, column l16
#pragma unroll
        for (int j = 0; j < 8; ++j) {
            b[j]     = brow[k0 + half * 8 + j];
            b[8 + j] = brow[k0 + 16 + half * 8 + j];
        }
        acc = __builtin_amdgcn_wmma_f32_16x16x32_f16(
            /*neg_a=*/false, a, /*neg_b=*/false, b,
            /*c_mod=*/(short)0, acc, /*reuse_a=*/false, /*reuse_b=*/false);
    }

    // D layout: vgpr r, lane L -> row M = r + 8*(L/16), col N = L%16
    const int ncol = tn * 16 + l16;
    const float bias = b_comb[ncol];
#pragma unroll
    for (int r = 0; r < 8; ++r) {
        int row = tm * 16 + r + half * 8;
        out[(size_t)row * kOut + ncol] = acc[r] + bias;
    }
}

// ---------------- launch ----------------
static inline size_t alignup(size_t x) { return (x + 511) & ~size_t(511); }

extern "C" void kernel_launch(void* const* d_in, const int* in_sizes, int n_in,
                              void* d_out, int out_size, void* d_ws, size_t ws_size,
                              hipStream_t stream) {
    const float* h_sent = (const float*)d_in[0];
    const float* h_type = (const float*)d_in[1];
    const float* attn_w = (const float*)d_in[2];
    const float* w_comb = (const float*)d_in[3];
    const float* b_comb = (const float*)d_in[4];
    const int*   src    = (const int*)d_in[5];
    const int*   dst    = (const int*)d_in[6];
    float* out = (float*)d_out;

    // carve workspace
    char* ws = (char*)d_ws;
    size_t off = 0;
    float* exbuf = (float*)(ws + off); off = alignup(off + (size_t)kEdges * kH * 4);
    float* m_buf = (float*)(ws + off); off = alignup(off + (size_t)kNType * kH * 4);
    float* den_buf = (float*)(ws + off); off = alignup(off + (size_t)kNType * kH * 4);
    float* h_accum = (float*)(ws + off); off = alignup(off + (size_t)kNType * kOut * 4);
    _Float16* h16 = (_Float16*)(ws + off); off = alignup(off + (size_t)kNType * kOut * 2);
    _Float16* w16 = (_Float16*)(ws + off); off = alignup(off + (size_t)kOut * kOut * 2);
    (void)ws_size; (void)n_in; (void)in_sizes; (void)out_size;

    const int B = 256;

    // pass 0: init (cover kNType*kOut elements)
    k_init<<<(kNType * kOut + B - 1) / B, B, 0, stream>>>(m_buf, den_buf, h_accum);

    // pass 1: logits + segment max  (E*H threads)
    long long nEH = (long long)kEdges * kH;
    k_logits<<<(int)((nEH + B - 1) / B), B, 0, stream>>>(
        h_sent, h_type, attn_w, src, dst, exbuf, m_buf);

    // pass 2: exp + segment sum
    k_expsum<<<(int)((nEH + B - 1) / B), B, 0, stream>>>(dst, m_buf, exbuf, den_buf);

    // pass 3: weighted scatter (E*256 threads)
    long long nEF = (long long)kEdges * kOut;
    k_scatter<<<(int)((nEF + B - 1) / B), B, 0, stream>>>(
        h_sent, src, dst, exbuf, den_buf, h_accum);

    // pass 4: convert to f16
    k_cvt<<<(kNType * kOut + B - 1) / B, B, 0, stream>>>(h_accum, w_comb, h16, w16);

    // pass 5: WMMA GEMM + bias  (10000 tiles, 8 waves/block)
    k_gemm<<<(kNType / 16) * 16 / 8, B, 0, stream>>>(h16, w16, b_comb, out);
}
